// MultiHeadSelfAttention_68307159876090
// MI455X (gfx1250) — matrix operand order repaired and checked by reading, not compiled, not run on hardware
//
#include <hip/hip_runtime.h>

// ---------------------------------------------------------------------------
// CDNA5 (gfx1250) fused multi-head self-attention
//   bf16 WMMA (v_wmma_f32_16x16x32_bf16) + f32 accumulate
//   Tensor Data Mover (tensor_load_to_lds) for K-tile streaming (guarded)
// ---------------------------------------------------------------------------

typedef __bf16 bf16_t;
typedef __attribute__((ext_vector_type(16))) __bf16        v16bf;
typedef __attribute__((ext_vector_type(8)))  float         v8f;
typedef __attribute__((ext_vector_type(4)))  unsigned int  u32x4;
typedef __attribute__((ext_vector_type(8)))  int           i32x8;
typedef __attribute__((ext_vector_type(4)))  int           i32x4;

#define D_MODEL   1024
#define NUM_HEADS 16
#define HEAD_DIM  64
#define SEQ       2048
#define BATCH     2
#define BS        (BATCH * SEQ)        // 4096 rows of X
#define BH        (BATCH * NUM_HEADS)  // 32

#if defined(__gfx1250__) && __has_builtin(__builtin_amdgcn_tensor_load_to_lds) && \
    __has_builtin(__builtin_amdgcn_s_wait_tensorcnt)
#define USE_TDM 1
#else
#define USE_TDM 0
#endif

// 32-byte fragment: 16 bf16 values, loadable as two 16B chunks
union Frag16 { v16bf v; u32x4 q[2]; };

__device__ __forceinline__ v8f wmma_bf16(v16bf a, v16bf b, v8f c) {
  // D(f32,16x16) = A(bf16,16x32) * B(bf16,32x16) + C
  return __builtin_amdgcn_wmma_f32_16x16x32_bf16(
      /*neg_a=*/false, a, /*neg_b=*/false, b,
      /*c_mod=*/(short)0, c, /*reuse_a=*/false, /*reuse_b=*/false);
}

#if USE_TDM
// TDM: DMA a 16-row x 64-elem bf16 tile (global row-major, 64-elem stride)
// into LDS. Descriptor per CDNA5 ISA. Wave-uniform.
__device__ __forceinline__ void tdm_load_k_tile(const bf16_t* gsrc,
                                                bf16_t* lds_dst) {
  unsigned long long ga = (unsigned long long)(size_t)gsrc;
  unsigned int ga_lo = (unsigned int)__builtin_amdgcn_readfirstlane((int)(unsigned int)ga);
  unsigned int ga_hi = (unsigned int)__builtin_amdgcn_readfirstlane((int)(unsigned int)(ga >> 32));
  // generic shared-aperture address: low 32 bits == LDS byte offset
  unsigned int lds   = (unsigned int)__builtin_amdgcn_readfirstlane((int)(unsigned int)(size_t)lds_dst);

  u32x4 g0;
  g0[0] = 1u;                                   // count=1, no gather
  g0[1] = lds;                                  // lds_addr
  g0[2] = ga_lo;                                // global_addr[31:0]
  g0[3] = (ga_hi & 0x01FFFFFFu) | 0x80000000u;  // addr[56:32] | type=2<<30

  i32x8 g1;
  g1[0] = 0x00010000;                // wg_mask=0, data_size=1 (2 bytes)
  g1[1] = (int)(64u << 16);          // tensor_dim0 = 64 (bits 79:48)
  g1[2] = (int)(2048u << 16);        // tensor_dim1 = 2048 (bits 111:80)
  g1[3] = (int)(64u << 16);          // tile_dim0 = 64 (bits 127:112)
  g1[4] = 16;                        // tile_dim1 = 16, tile_dim2 = 0
  g1[5] = 64;                        // tensor_dim0_stride = 64
  g1[6] = 0;
  g1[7] = 0;

  i32x4 z4 = {0, 0, 0, 0};
#if __has_include(<hip/amd_detail/amd_gfx1250_TDM.h>)
  i32x8 z8 = {0, 0, 0, 0, 0, 0, 0, 0};
  __builtin_amdgcn_tensor_load_to_lds(g0, g1, z4, z4, z8, 0);
#else
  __builtin_amdgcn_tensor_load_to_lds(g0, g1, z4, z4, 0);
#endif
}
#endif  // USE_TDM

// ---------------------------------------------------------------------------
// fp32 -> bf16 conversion (grid-stride)
// ---------------------------------------------------------------------------
__global__ void cvt_f32_bf16_kernel(const float* __restrict__ src,
                                    bf16_t* __restrict__ dst, int n) {
  int i = blockIdx.x * blockDim.x + threadIdx.x;
  int stride = gridDim.x * blockDim.x;
  for (; i < n; i += stride) dst[i] = (bf16_t)src[i];
}

// ---------------------------------------------------------------------------
// GEMM: Y[m,e] = sum_k A[m,k] * W[e,k] + bias[e]
//   A: [4096,1024] bf16 row-major     W: [1024,1024] bf16 row-major (torch W)
// mode 0: store bf16 [BH, S, HEAD_DIM]      (Q, K)
// mode 1: store bf16 [BH, HEAD_DIM, S]      (V transposed)
// mode 2: store f32  [BS, D_MODEL]          (final output)
// One 16x64 strip per wave (A-fragment reused across 4 WMMAs); 8 waves/block.
// ---------------------------------------------------------------------------
__global__ void gemm_proj_kernel(const bf16_t* __restrict__ A,
                                 const bf16_t* __restrict__ W,
                                 const float*  __restrict__ bias,
                                 void* __restrict__ outp, int mode) {
  const int lane = threadIdx.x & 31;
  const int wave = threadIdx.x >> 5;
  const int h    = lane >> 4;   // half-wave index
  const int ln   = lane & 15;

  const int tile = blockIdx.x * 8 + wave;  // 4096 wave-strips total
  const int mt = tile >> 4;                // M tile (0..255)
  const int ng = tile & 15;                // N group of 64 (0..15)
  const int m0 = mt * 16, n0 = ng * 64;

  const bf16_t* arow = A + (size_t)(m0 + ln) * D_MODEL;  // A-frag row
  const bf16_t* wrow[4];
  #pragma unroll
  for (int j = 0; j < 4; ++j)
    wrow[j] = W + (size_t)(n0 + j * 16 + ln) * D_MODEL + h * 16;

  v8f acc[4];
  #pragma unroll
  for (int j = 0; j < 4; ++j)
    acc[j] = v8f{0.f, 0.f, 0.f, 0.f, 0.f, 0.f, 0.f, 0.f};

  #pragma unroll 2
  for (int k0 = 0; k0 < D_MODEL; k0 += 32) {
    Frag16 a;
    // A-fragment (16x32): lane holds row m0+ln, K contiguous from 16h
    a.q[0] = *reinterpret_cast<const u32x4*>(arow + k0 + h * 16);
    a.q[1] = *reinterpret_cast<const u32x4*>(arow + k0 + h * 16 + 8);
    __builtin_prefetch(arow + k0 + 256, 0, 1);   // global_prefetch_b8
    #pragma unroll
    for (int j = 0; j < 4; ++j) {
      Frag16 b;  // B-fragment (32x16): lane holds col, K contiguous from 16h
      const u32x4* wp = reinterpret_cast<const u32x4*>(wrow[j] + k0);
      b.q[0] = wp[0];
      b.q[1] = wp[1];
      acc[j] = wmma_bf16(a.v, b.v, acc[j]);
    }
  }

  #pragma unroll
  for (int j = 0; j < 4; ++j) {
    const float bv = bias[n0 + j * 16 + ln];
    #pragma unroll
    for (int r = 0; r < 8; ++r) {
      const int m = m0 + r + 8 * h;     // D-tile: VGPR r, half h -> row
      const int e = n0 + j * 16 + ln;   // col = lane%16
      const float val = acc[j][r] + bv;
      if (mode == 2) {
        ((float*)outp)[(size_t)m * D_MODEL + e] = val;
      } else {
        const int bb = m >> 11;          // / SEQ
        const int s  = m & (SEQ - 1);
        const int head = e >> 6;
        const int hd   = e & 63;
        bf16_t* o = (bf16_t*)outp;
        if (mode == 0)
          o[(((size_t)(bb * NUM_HEADS + head)) * SEQ + s) * HEAD_DIM + hd] =
              (bf16_t)val;
        else
          o[(((size_t)(bb * NUM_HEADS + head)) * HEAD_DIM + hd) * SEQ + s] =
              (bf16_t)val;
      }
    }
  }
}

// ---------------------------------------------------------------------------
// Causal attention: scores=QK^T/8, softmax, write attn (f32), O=P*V (bf16)
// grid = (SEQ/128, BH), block = 256 (8 waves); wave w owns q rows
// [blockIdx.x*128 + w*16, +16). Pass 1 = online stats (K tiles streamed via
// TDM double-buffer when available); pass 2 = emit attn + P*V.
// ---------------------------------------------------------------------------
#define PRS 48   // LDS row stride for P staging (bf16 elems); 96B

__global__ void attn_kernel(const bf16_t* __restrict__ Qb,
                            const bf16_t* __restrict__ Kb,
                            const bf16_t* __restrict__ Vt,
                            float* __restrict__ attn,
                            bf16_t* __restrict__ Ab) {
  __shared__ __align__(16) bf16_t Psh[8 * 16 * PRS];
#if USE_TDM
  __shared__ __align__(16) bf16_t Ksh[8][2][16 * HEAD_DIM];  // 32 KB
#endif

  const int lane = threadIdx.x & 31;
  const int wave = threadIdx.x >> 5;
  const int h  = lane >> 4;
  const int ln = lane & 15;
  const int bh = blockIdx.y;
  const int b    = bh >> 4;
  const int head = bh & 15;
  const int q0   = blockIdx.x * 128 + wave * 16;
  const int numk = q0 / 16 + 1;                 // causal: k tiles 0..q0/16

  const bf16_t* Qbase = Qb + ((size_t)bh * SEQ + q0) * HEAD_DIM;
  const bf16_t* Kbase = Kb + (size_t)bh * SEQ * HEAD_DIM;
  const bf16_t* Vbase = Vt + (size_t)bh * HEAD_DIM * SEQ;
  float* attnBase = attn + (size_t)bh * SEQ * SEQ;
  bf16_t* Psw = Psh + wave * 16 * PRS;

  // Q A-fragments: hd 0..31 and 32..63 (lane holds row q0+ln)
  Frag16 aq0, aq1;
  {
    const bf16_t* qr = Qbase + (size_t)ln * HEAD_DIM;
    aq0.q[0] = *reinterpret_cast<const u32x4*>(qr + h * 16);
    aq0.q[1] = *reinterpret_cast<const u32x4*>(qr + h * 16 + 8);
    aq1.q[0] = *reinterpret_cast<const u32x4*>(qr + 32 + h * 16);
    aq1.q[1] = *reinterpret_cast<const u32x4*>(qr + 32 + h * 16 + 8);
  }

  const float NEGINF = -__builtin_inff();
  float M[8], Ssum[8];
  #pragma unroll
  for (int r = 0; r < 8; ++r) { M[r] = NEGINF; Ssum[r] = 0.f; }

  // ---------------- pass 1: online softmax statistics ----------------
#if USE_TDM
  tdm_load_k_tile(Kbase, &Ksh[wave][0][0]);   // prefetch tile 0
#endif
  for (int kt = 0; kt < numk; ++kt) {
    Frag16 b0, b1;   // B = K^T: col = K row (contiguous hd)
#if USE_TDM
    if (kt + 1 < numk) {
      tdm_load_k_tile(Kbase + (size_t)(kt + 1) * 16 * HEAD_DIM,
                      &Ksh[wave][(kt + 1) & 1][0]);
      __builtin_amdgcn_s_wait_tensorcnt(1);   // current buffer resident
    } else {
      __builtin_amdgcn_s_wait_tensorcnt(0);
    }
    const bf16_t* kr = &Ksh[wave][kt & 1][0] + (size_t)ln * HEAD_DIM;
#else
    const bf16_t* kr = Kbase + (size_t)(kt * 16 + ln) * HEAD_DIM;
#endif
    b0.q[0] = *reinterpret_cast<const u32x4*>(kr + h * 16);
    b0.q[1] = *reinterpret_cast<const u32x4*>(kr + h * 16 + 8);
    b1.q[0] = *reinterpret_cast<const u32x4*>(kr + 32 + h * 16);
    b1.q[1] = *reinterpret_cast<const u32x4*>(kr + 32 + h * 16 + 8);
    v8f s = {0.f, 0.f, 0.f, 0.f, 0.f, 0.f, 0.f, 0.f};
    s = wmma_bf16(aq0.v, b0.v, s);
    s = wmma_bf16(aq1.v, b1.v, s);

    const int kc = kt * 16 + ln;
    #pragma unroll
    for (int r = 0; r < 8; ++r) {
      const int q = q0 + r + 8 * h;
      float sv = (kc <= q) ? s[r] * 0.125f : NEGINF;   // 1/sqrt(64)
      float rm = sv;   // row max over the 16 lanes holding this row
      rm = fmaxf(rm, __shfl_xor(rm, 1, 32));
      rm = fmaxf(rm, __shfl_xor(rm, 2, 32));
      rm = fmaxf(rm, __shfl_xor(rm, 4, 32));
      rm = fmaxf(rm, __shfl_xor(rm, 8, 32));
      const float nm = fmaxf(M[r], rm);
      float pe = (sv > NEGINF) ? __expf(sv - nm) : 0.f;
      float rs = pe;
      rs += __shfl_xor(rs, 1, 32);
      rs += __shfl_xor(rs, 2, 32);
      rs += __shfl_xor(rs, 4, 32);
      rs += __shfl_xor(rs, 8, 32);
      const float alpha = (M[r] > NEGINF) ? __expf(M[r] - nm) : 0.f;
      Ssum[r] = Ssum[r] * alpha + rs;
      M[r] = nm;
    }
  }

  float invS[8];
  #pragma unroll
  for (int r = 0; r < 8; ++r) invS[r] = (Ssum[r] > 0.f) ? 1.f / Ssum[r] : 0.f;

  // ---------------- pass 2: emit attn and accumulate O = P*V ----------------
  v8f o[4];
  #pragma unroll
  for (int i = 0; i < 4; ++i)
    o[i] = v8f{0.f, 0.f, 0.f, 0.f, 0.f, 0.f, 0.f, 0.f};

  const int npairs = (numk + 1) >> 1;
  for (int t = 0; t < npairs; ++t) {
    #pragma unroll
    for (int u = 0; u < 2; ++u) {
      const int kt = 2 * t + u;
      if (kt < numk) {
        const bf16_t* kr = Kbase + (size_t)(kt * 16 + ln) * HEAD_DIM;
        Frag16 b0, b1;
        b0.q[0] = *reinterpret_cast<const u32x4*>(kr + h * 16);
        b0.q[1] = *reinterpret_cast<const u32x4*>(kr + h * 16 + 8);
        b1.q[0] = *reinterpret_cast<const u32x4*>(kr + 32 + h * 16);
        b1.q[1] = *reinterpret_cast<const u32x4*>(kr + 32 + h * 16 + 8);
        v8f s = {0.f, 0.f, 0.f, 0.f, 0.f, 0.f, 0.f, 0.f};
        s = wmma_bf16(aq0.v, b0.v, s);
        s = wmma_bf16(aq1.v, b1.v, s);
        const int kc = kt * 16 + ln;
        #pragma unroll
        for (int r = 0; r < 8; ++r) {
          const int q = q0 + r + 8 * h;
          const float p =
              (kc <= q) ? __expf(s[r] * 0.125f - M[r]) * invS[r] : 0.f;
          attnBase[(size_t)q * SEQ + kc] = p;                    // f32 attn
          Psw[(r + 8 * h) * PRS + u * 16 + ln] = (bf16_t)p;      // stage
        }
      } else {
        #pragma unroll
        for (int r = 0; r < 8; ++r)
          Psw[(r + 8 * h) * PRS + u * 16 + ln] = (bf16_t)0.f;
      }
    }
    // A-fragment of P (16 q x 32 k) from this wave's LDS region
    Frag16 ap;
    const bf16_t* pr = Psw + ln * PRS;
    ap.q[0] = *reinterpret_cast<const u32x4*>(pr + h * 16);
    ap.q[1] = *reinterpret_cast<const u32x4*>(pr + h * 16 + 8);
    #pragma unroll
    for (int n4 = 0; n4 < 4; ++n4) {
      // B = V chunk (32 k x 16 hd); V^T layout makes columns contiguous
      const bf16_t* vr =
          Vbase + (size_t)(n4 * 16 + ln) * SEQ + t * 32 + h * 16;
      Frag16 bv;
      bv.q[0] = *reinterpret_cast<const u32x4*>(vr);
      bv.q[1] = *reinterpret_cast<const u32x4*>(vr + 8);
      o[n4] = wmma_bf16(ap.v, bv.v, o[n4]);
    }
  }

  // zero-fill the masked (upper-triangle) attn tiles
  for (int kt = numk; kt < SEQ / 16; ++kt) {
    const int kc = kt * 16 + ln;
    #pragma unroll
    for (int r = 0; r < 8; ++r)
      attnBase[(size_t)(q0 + r + 8 * h) * SEQ + kc] = 0.f;
  }

  // write O tile as bf16 into A buffer [B, S, D]
  #pragma unroll
  for (int n4 = 0; n4 < 4; ++n4) {
    #pragma unroll
    for (int r = 0; r < 8; ++r) {
      const int q  = q0 + r + 8 * h;
      const int hd = n4 * 16 + ln;
      Ab[((size_t)b * SEQ + q) * D_MODEL + head * HEAD_DIM + hd] =
          (bf16_t)o[n4][r];
    }
  }
}

// ---------------------------------------------------------------------------
// host-side orchestration
// ---------------------------------------------------------------------------
extern "C" void kernel_launch(void* const* d_in, const int* in_sizes, int n_in,
                              void* d_out, int out_size, void* d_ws,
                              size_t ws_size, hipStream_t stream) {
  const float* X  = (const float*)d_in[0];
  const float* Wq = (const float*)d_in[1];
  const float* bq = (const float*)d_in[2];
  const float* Wk = (const float*)d_in[3];
  const float* bk = (const float*)d_in[4];
  const float* Wv = (const float*)d_in[5];
  const float* bv = (const float*)d_in[6];
  const float* Wo = (const float*)d_in[7];
  const float* bo = (const float*)d_in[8];

  // workspace layout (bf16 elements), ~48 MiB total
  bf16_t* Xb  = (bf16_t*)d_ws;              // 4096*1024
  bf16_t* Wqb = Xb  + (size_t)BS * D_MODEL;
  bf16_t* Wkb = Wqb + (size_t)D_MODEL * D_MODEL;
  bf16_t* Wvb = Wkb + (size_t)D_MODEL * D_MODEL;
  bf16_t* Wob = Wvb + (size_t)D_MODEL * D_MODEL;
  bf16_t* Qb  = Wob + (size_t)D_MODEL * D_MODEL;    // [BH,S,HD]
  bf16_t* Kb  = Qb  + (size_t)BH * SEQ * HEAD_DIM;  // [BH,S,HD]
  bf16_t* Vt  = Kb  + (size_t)BH * SEQ * HEAD_DIM;  // [BH,HD,S]
  bf16_t* Ab  = Vt  + (size_t)BH * SEQ * HEAD_DIM;  // [B,S,D]

  float* out  = (float*)d_out;                       // [B,S,D]
  float* attn = out + (size_t)BS * D_MODEL;          // [B,H,S,S]

  // 1) convert inputs to bf16
  cvt_f32_bf16_kernel<<<2048, 256, 0, stream>>>(X,  Xb,  BS * D_MODEL);
  cvt_f32_bf16_kernel<<<1024, 256, 0, stream>>>(Wq, Wqb, D_MODEL * D_MODEL);
  cvt_f32_bf16_kernel<<<1024, 256, 0, stream>>>(Wk, Wkb, D_MODEL * D_MODEL);
  cvt_f32_bf16_kernel<<<1024, 256, 0, stream>>>(Wv, Wvb, D_MODEL * D_MODEL);
  cvt_f32_bf16_kernel<<<1024, 256, 0, stream>>>(Wo, Wob, D_MODEL * D_MODEL);

  // 2) projections (Q,K as [BH,S,HD]; V transposed to [BH,HD,S])
  const int gemm_blocks = (BS / 16) * (D_MODEL / 64) / 8;  // 512
  gemm_proj_kernel<<<gemm_blocks, 256, 0, stream>>>(Xb, Wqb, bq, Qb, 0);
  gemm_proj_kernel<<<gemm_blocks, 256, 0, stream>>>(Xb, Wkb, bk, Kb, 0);
  gemm_proj_kernel<<<gemm_blocks, 256, 0, stream>>>(Xb, Wvb, bv, Vt, 1);

  // 3) causal attention: attn (f32) + O (bf16)
  attn_kernel<<<dim3(SEQ / 128, BH), 256, 0, stream>>>(Qb, Kb, Vt, attn, Ab);

  // 4) output projection (f32 result + bias)
  gemm_proj_kernel<<<gemm_blocks, 256, 0, stream>>>(Ab, Wob, bo, out, 2);
}
